// _UniMatchStereoNet_28509992911257
// MI455X (gfx1250) — matrix operand order repaired and checked by reading, compiled, not run on hardware
//
#include <hip/hip_runtime.h>
#include <hip/hip_bf16.h>

// Shifted-window attention (UniMatch) for MI455X / gfx1250.
// Reference config: B=8, H=96, W=128, C=128, NS=4
// -> 128 windows x 768 tokens x head-dim 128.
// Flash-attention per 16-row q tile using v_wmma_f32_16x16x32_f16 (wave32).
// Softmax row reductions use DPP16 butterflies (pure VALU, no ds_bpermute).

typedef __attribute__((ext_vector_type(16))) _Float16 v16h;
typedef __attribute__((ext_vector_type(8)))  _Float16 v8h;
typedef __attribute__((ext_vector_type(8)))  float    v8f;

#define HH    96
#define WW    128
#define CC    128
#define WSH   24
#define WSW   32
#define NTOK  768           // WSH*WSW tokens per window
#define SHH   12            // shift = WSH/2
#define SWW   16            // shift = WSW/2
#define KSTR  136           // K LDS row stride (halves): 128 + 8 pad
#define VTSTR 40            // V^T LDS row stride (halves): 32 + 8 pad
#define PSTR  40            // P buffer row stride (halves)

// roll(+sh,+sw) applied to window token t -> flat image index. Input gather
// and output scatter both use this (input roll and inverse output roll on the
// merged grid are the same coordinate map).
__device__ __forceinline__ int rolled_flat(int t, int sy, int sx, int shift) {
  int y = t >> 5, x = t & 31;            // within-window coords (WSW = 32)
  int hh = sy * WSH + y + (shift ? SHH : 0);
  int ww = sx * WSW + x + (shift ? SWW : 0);
  if (hh >= HH) hh -= HH;
  if (ww >= WW) ww -= WW;
  return hh * WW + ww;
}

// ---- DPP16 cross-lane moves (VALU only; rows of 16 lanes) ----
template <int CTRL>
__device__ __forceinline__ float dpp_movf(float x) {
  return __builtin_bit_cast(float,
      __builtin_amdgcn_update_dpp(0, __builtin_bit_cast(int, x),
                                  CTRL, 0xF, 0xF, true));
}
// Full 16-lane butterfly reductions (per DPP row; wave32 = two rows).
__device__ __forceinline__ float red16_max(float x) {
  x = fmaxf(x, dpp_movf<0xB1>(x));    // quad_perm(1,0,3,2): xor 1
  x = fmaxf(x, dpp_movf<0x4E>(x));    // quad_perm(2,3,0,1): xor 2
  x = fmaxf(x, dpp_movf<0x141>(x));   // ROW_HALF_MIRROR: cross quads in 8-half
  x = fmaxf(x, dpp_movf<0x140>(x));   // ROW_MIRROR: cross the two 8-halves
  return x;
}
__device__ __forceinline__ float red16_sum(float x) {
  x += dpp_movf<0xB1>(x);
  x += dpp_movf<0x4E>(x);
  x += dpp_movf<0x141>(x);
  x += dpp_movf<0x140>(x);
  return x;
}

// Build a WMMA A-fragment lane vector from two 8-float global segments.
__device__ __forceinline__ v16h ld_g2x8(const float* __restrict__ p0,
                                        const float* __restrict__ p1) {
  v16h r;
#pragma unroll
  for (int i = 0; i < 8; ++i) { r[i] = (_Float16)p0[i]; r[i + 8] = (_Float16)p1[i]; }
  return r;
}

// 16 contiguous halves from LDS (16-byte aligned) -> B-fragment lane vector.
__device__ __forceinline__ v16h ld_lds16(const _Float16* p) {
  v8h a = *(const v8h*)p;
  v8h b = *(const v8h*)(p + 8);
  v16h r;
#pragma unroll
  for (int i = 0; i < 8; ++i) { r[i] = a[i]; r[i + 8] = b[i]; }
  return r;
}

// Two 8-half LDS segments -> A-fragment lane vector.
__device__ __forceinline__ v16h ld_lds2x8(const _Float16* p0, const _Float16* p1) {
  v8h a = *(const v8h*)p0;
  v8h b = *(const v8h*)p1;
  v16h r;
#pragma unroll
  for (int i = 0; i < 8; ++i) { r[i] = a[i]; r[i + 8] = b[i]; }
  return r;
}

__global__ __launch_bounds__(256)
void swin_flash_attn_kernel(const float* __restrict__ q,
                            const float* __restrict__ k,
                            const float* __restrict__ v,
                            const float* __restrict__ mask,
                            const int*   __restrict__ shift_p,
                            float* __restrict__ out) {
  __shared__ __align__(32) _Float16 Klds[32 * KSTR];      // K chunk, row-major [key][c]
  __shared__ __align__(32) _Float16 VTlds[CC * VTSTR];    // V chunk transposed [c][key]
  __shared__ __align__(32) _Float16 Pbuf[8 * 16 * PSTR];  // per-wave P tile bounce

  const int shift = shift_p[0];
  const int widx  = blockIdx.x / 6;       // window 0..127
  const int qblk  = blockIdx.x % 6;       // 128-row slab within window
  const int b     = widx >> 4;
  const int wi    = widx & 15;            // mask index (tile repeats over batch)
  const int sy    = wi >> 2, sx = wi & 3;

  const int tid  = threadIdx.x;
  const int wave = tid >> 5;
  const int lane = tid & 31;
  const int ln   = lane & 15;             // N index / row-in-16
  const int hi   = lane >> 4;             // lane half

  const size_t bOff   = (size_t)b * (HH * WW) * CC;
  const int    qwbase = qblk * 128 + wave * 16;   // first q row of this wave's tile
  const float* mbase  = mask + (size_t)wi * NTOK * NTOK;

  // Cooperative-load role of this thread (fixed across chunks).
  const int ld_i  = tid >> 3;             // key 0..31 within chunk
  const int ld_c0 = (tid & 7) << 4;       // c base 0..112

  // ---- Q tile (16 x 128) in WMMA A layout: 4 fragments over the c dim ----
  v16h aQ[4];
  {
    const int    qflat = rolled_flat(qwbase + ln, sy, sx, shift);
    const float* qrow  = q + bOff + (size_t)qflat * CC;
#pragma unroll
    for (int kc = 0; kc < 4; ++kc) {
      const float* p0 = qrow + kc * 32 + 16 * hi;  // A layout: k = 16*hi + 0..7
      aQ[kc] = ld_g2x8(p0, p0 + 8);                //           k = 16*hi + 8 + 0..7
    }
  }

  const v8f zero8 = {0.f, 0.f, 0.f, 0.f, 0.f, 0.f, 0.f, 0.f};
  v8f   acc[8];                                    // O accumulator: 8 c-tiles (16x128)
  float mrun[8], lrun[8];
#pragma unroll
  for (int j = 0; j < 8; ++j) { mrun[j] = -1e30f; lrun[j] = 0.f; }
#pragma unroll
  for (int t = 0; t < 8; ++t) acc[t] = zero8;

  const float sc = 0.088388347648318447f;          // 1/sqrt(128)

  for (int kb = 0; kb < NTOK; kb += 32) {
    // ---- cooperative K/V chunk load: 32 keys x 128 c, f32 -> f16 ----
    {
      const int    kflat = rolled_flat(kb + ld_i, sy, sx, shift);
      const float* kr    = k + bOff + (size_t)kflat * CC + ld_c0;
      const float* vr    = v + bOff + (size_t)kflat * CC + ld_c0;
      _Float16*    kd    = Klds + ld_i * KSTR + ld_c0;
      v8h h0, h1;
#pragma unroll
      for (int e = 0; e < 8; ++e) { h0[e] = (_Float16)kr[e]; h1[e] = (_Float16)kr[8 + e]; }
      *(v8h*)kd       = h0;                        // ds_store_b128
      *(v8h*)(kd + 8) = h1;                        // ds_store_b128
#pragma unroll
      for (int e = 0; e < 16; ++e) VTlds[(ld_c0 + e) * VTSTR + ld_i] = (_Float16)vr[e];
    }
    __syncthreads();

    // Prefetch next chunk's K/V rows while we compute on this one.
    if (kb + 32 < NTOK) {
      const int nflat = rolled_flat(kb + 32 + ld_i, sy, sx, shift);
      __builtin_prefetch(k + bOff + (size_t)nflat * CC + ld_c0, 0, 0);
      __builtin_prefetch(v + bOff + (size_t)nflat * CC + ld_c0, 0, 0);
    }

    // ---- S = Q K^T for this 32-key chunk: two 16x16 f32 tiles ----
    v8f s0 = zero8, s1 = zero8;
#pragma unroll
    for (int kc = 0; kc < 4; ++kc) {
      const int c0 = kc * 32 + 16 * hi;            // B layout: k = 16*hi + 0..15
      v16h b0 = ld_lds16(Klds + ln * KSTR + c0);         // keys kb+0..15
      v16h b1 = ld_lds16(Klds + (16 + ln) * KSTR + c0);  // keys kb+16..31
      s0 = __builtin_amdgcn_wmma_f32_16x16x32_f16(false, aQ[kc], false, b0,
                                                  (short)0, s0, false, false);
      s1 = __builtin_amdgcn_wmma_f32_16x16x32_f16(false, aQ[kc], false, b1,
                                                  (short)0, s1, false, false);
    }

    // ---- online softmax update (rows: j + 8*hi; cols striped on lanes) ----
    float alpha[8], p0v[8], p1v[8];
#pragma unroll
    for (int j = 0; j < 8; ++j) {
      float a0 = s0[j] * sc;
      float a1 = s1[j] * sc;
      if (shift) {
        const int r = qwbase + j + 8 * hi;
        a0 += mbase[(size_t)r * NTOK + kb + ln];
        a1 += mbase[(size_t)r * NTOK + kb + 16 + ln];
      }
      const float tmax = red16_max(fmaxf(a0, a1));
      const float mnew = fmaxf(mrun[j], tmax);
      const float e0 = __expf(a0 - mnew);
      const float e1 = __expf(a1 - mnew);
      const float rs = red16_sum(e0 + e1);
      alpha[j] = __expf(mrun[j] - mnew);
      lrun[j]  = lrun[j] * alpha[j] + rs;
      mrun[j]  = mnew;
      p0v[j] = e0; p1v[j] = e1;
    }
#pragma unroll
    for (int t = 0; t < 8; ++t) {
#pragma unroll
      for (int j = 0; j < 8; ++j) acc[t][j] *= alpha[j];
    }

    // ---- P: D layout -> LDS -> A layout; then O += P * V ----
    {
      _Float16* pb = Pbuf + wave * 16 * PSTR;
#pragma unroll
      for (int j = 0; j < 8; ++j) {
        const int r = j + 8 * hi;
        pb[r * PSTR + ln]      = (_Float16)p0v[j];
        pb[r * PSTR + 16 + ln] = (_Float16)p1v[j];
      }
      asm volatile("s_wait_dscnt 0x0" ::: "memory");   // wave-local WAR/RAW guard
      v16h aP = ld_lds2x8(pb + ln * PSTR + 16 * hi,
                          pb + ln * PSTR + 16 * hi + 8);
#pragma unroll
      for (int t = 0; t < 8; ++t) {
        v16h bv = ld_lds16(VTlds + (t * 16 + ln) * VTSTR + 16 * hi);
        acc[t] = __builtin_amdgcn_wmma_f32_16x16x32_f16(false, aP, false, bv,
                                                        (short)0, acc[t], false, false);
      }
    }
    __syncthreads();   // protect K/V LDS before next chunk overwrites
  }

  // ---- normalize and scatter O (rolled coordinates fold the inverse roll) ----
  float inv[8];
  int   oflat[8];
#pragma unroll
  for (int j = 0; j < 8; ++j) {
    inv[j]   = 1.0f / lrun[j];
    oflat[j] = rolled_flat(qwbase + j + 8 * hi, sy, sx, shift);
  }
#pragma unroll
  for (int t = 0; t < 8; ++t) {
    const int c = t * 16 + ln;
#pragma unroll
    for (int j = 0; j < 8; ++j) {
      out[bOff + (size_t)oflat[j] * CC + c] = acc[t][j] * inv[j];
    }
  }
}

extern "C" void kernel_launch(void* const* d_in, const int* in_sizes, int n_in,
                              void* d_out, int out_size, void* d_ws, size_t ws_size,
                              hipStream_t stream) {
  const float* q    = (const float*)d_in[0];
  const float* k    = (const float*)d_in[1];
  const float* v    = (const float*)d_in[2];
  const float* mask = (const float*)d_in[3];
  const int*   wsh  = (const int*)d_in[5];   // with_shift scalar (device)
  float*       out  = (float*)d_out;

  const int B = in_sizes[0] / (HH * WW * CC);          // 8 for the reference setup
  dim3 grid(B * 16 * 6);                               // windows * q-slabs
  dim3 block(256);                                     // 8 wave32 per block
  hipLaunchKernelGGL(swin_flash_attn_kernel, grid, block, 0, stream,
                     q, k, v, mask, wsh, out);
}
